// DeformableConvBlock_75900662055626
// MI455X (gfx1250) — matrix operand order repaired and checked by reading, not compiled, not run on hardware
//
#include <hip/hip_runtime.h>
#include <cstdint>
#include <cstddef>

// ---------------------------------------------------------------------------
// MI455X (gfx1250) deformable-conv block, bf16 WMMA pipeline.
// D = A(16x32 bf16) x B(32x16 bf16) + C(16x16 f32) via v_wmma_f32_16x16x32_bf16
// B tiles staged with GLOBAL_LOAD_ASYNC_TO_LDS_B128 (ASYNCcnt) on CDNA5.
// ---------------------------------------------------------------------------

typedef __attribute__((ext_vector_type(16))) __bf16 v16bf;
typedef __attribute__((ext_vector_type(8)))  float  v8f;
typedef __attribute__((ext_vector_type(4)))  int    i4;

#define Bb   8
#define Cc   128
#define Oo   256
#define Hh   64
#define Ww   64
#define HW   4096
#define K2T  9
#define CK   1152          // C * 9
#define NKC  36            // CK / 32 k-chunks for the big GEMM

#if defined(__has_builtin)
#if __has_builtin(__builtin_amdgcn_global_load_async_to_lds_b128)
#define HAVE_ASYNC_LDS 1
#endif
#if __has_builtin(__builtin_amdgcn_s_wait_asynccnt)
#define HAVE_WAIT_ASYNC 1
#endif
#endif

#ifdef HAVE_ASYNC_LDS
typedef __attribute__((address_space(1))) i4 gi4;   // global int4
typedef __attribute__((address_space(3))) i4 li4;   // LDS int4
#endif

union AF { unsigned u[8]; v16bf v; };

__device__ __forceinline__ unsigned pack_bf16(float a, float b) {
  __bf16 ha = (__bf16)a, hb = (__bf16)b;
  unsigned short ba = __builtin_bit_cast(unsigned short, ha);
  unsigned short bb = __builtin_bit_cast(unsigned short, hb);
  return (unsigned)ba | ((unsigned)bb << 16);
}

// Stage a 16 KB B tile (4096 u32) global -> LDS with 256 threads.
// Uses CDNA5 async direct-to-LDS path when the builtin exists; otherwise
// regular b128 load + ds_store. Caller must __syncthreads() afterwards.
__device__ __forceinline__ void stage_b_tile(const unsigned* __restrict__ src,
                                             unsigned* dstLds, int tid) {
#ifdef HAVE_ASYNC_LDS
#pragma unroll
  for (int i = 0; i < 4; ++i) {
    __builtin_amdgcn_global_load_async_to_lds_b128(
        (gi4*)(src + (tid + 256 * i) * 4),
        (li4*)(dstLds + (tid + 256 * i) * 4),
        0, 0);
  }
#ifdef HAVE_WAIT_ASYNC
  __builtin_amdgcn_s_wait_asynccnt(0);
#else
  asm volatile("s_wait_asynccnt 0" ::: "memory");
#endif
#else
  const uint4* s4 = (const uint4*)src;
  uint4* d4 = (uint4*)dstLds;
#pragma unroll
  for (int i = 0; i < 4; ++i) d4[tid + 256 * i] = s4[tid + 256 * i];
#endif
}

// ---------------------------------------------------------------------------
// prep: BN folding + weight repack into bf16 B-fragment layout.
// B fragment (32x16, K x N): lane 0-15 -> K=0..15 (VGPR v holds K=2v,2v+1),
// lane 16-31 -> K=16..31.   (this code's contiguous k order, not the ISA table order)
// ---------------------------------------------------------------------------
__global__ __launch_bounds__(256) void prep_kernel(
    const float* __restrict__ w_off, const float* __restrict__ w_mod,
    const float* __restrict__ w_reg, const float* __restrict__ sc_w,
    const float* __restrict__ b_reg,
    const float* __restrict__ g1, const float* __restrict__ be1,
    const float* __restrict__ mu1, const float* __restrict__ va1,
    const float* __restrict__ g2, const float* __restrict__ be2,
    const float* __restrict__ mu2, const float* __restrict__ va2,
    unsigned* __restrict__ wregB, unsigned* __restrict__ woffB,
    unsigned* __restrict__ wscB, float* __restrict__ bnp)
{
  int gid = blockIdx.x * blockDim.x + threadIdx.x;

  if (gid < Oo) {
    float inv1 = g1[gid] * rsqrtf(va1[gid] + 1e-5f);
    bnp[gid]        = inv1;
    bnp[256 + gid]  = (b_reg[gid] - mu1[gid]) * inv1 + be1[gid];
    float inv2 = g2[gid] * rsqrtf(va2[gid] + 1e-5f);
    bnp[512 + gid]  = inv2;
    bnp[768 + gid]  = be2[gid] - mu2[gid] * inv2;
  }

  const int N_REG = 147456;            // 36 kc * 16 nt * 32 lane * 8 v
  const int N_OFF = 18432;             // 36 kc * 2  nt * 32 lane * 8 v
  const int N_SC  = 16384;             // 4  kc * 16 nt * 32 lane * 8 v
  if (gid >= N_REG + N_OFF + N_SC) return;

  if (gid < N_REG) {
    int fi = gid;
    int kc = fi >> 12;  int rem  = fi & 4095;
    int nt = rem >> 8;  int rem2 = rem & 255;
    int lane = rem2 >> 3; int v = rem2 & 7;
    int N  = nt * 16 + (lane & 15);
    int Kb = kc * 32 + ((lane >> 4) << 4) + 2 * v;
    int t0 = Kb / Cc,  c0 = Kb % Cc;
    int t1 = (Kb + 1) / Cc, c1 = (Kb + 1) % Cc;
    float f0 = w_reg[(N * Cc + c0) * K2T + t0];
    float f1 = w_reg[(N * Cc + c1) * K2T + t1];
    wregB[((kc * 16 + nt) * 32 + lane) * 8 + v] = pack_bf16(f0, f1);
  } else if (gid < N_REG + N_OFF) {
    int fi = gid - N_REG;
    int kc = fi >> 9;   int rem  = fi & 511;
    int nt = rem >> 8;  int rem2 = rem & 255;
    int lane = rem2 >> 3; int v = rem2 & 7;
    int N  = nt * 16 + (lane & 15);
    int Kb = kc * 32 + ((lane >> 4) << 4) + 2 * v;
    float f[2];
#pragma unroll
    for (int h = 0; h < 2; ++h) {
      int K = Kb + h; int t = K / Cc, c = K % Cc;
      float val = 0.f;
      if (N < 18)      val = w_off[(N * Cc + c) * K2T + t];
      else if (N < 27) val = w_mod[((N - 18) * Cc + c) * K2T + t];
      f[h] = val;
    }
    woffB[((kc * 2 + nt) * 32 + lane) * 8 + v] = pack_bf16(f[0], f[1]);
  } else {
    int fi = gid - N_REG - N_OFF;
    int kc = fi >> 12;  int rem  = fi & 4095;
    int nt = rem >> 8;  int rem2 = rem & 255;
    int lane = rem2 >> 3; int v = rem2 & 7;
    int N  = nt * 16 + (lane & 15);
    int Kb = kc * 32 + ((lane >> 4) << 4) + 2 * v;
    float f0 = sc_w[N * Cc + Kb];
    float f1 = sc_w[N * Cc + Kb + 1];
    wscB[((kc * 16 + nt) * 32 + lane) * 8 + v] = pack_bf16(f0, f1);
  }
}

// A fragment (16x32, M x K) bf16: lane 0-15 M=lane, VGPR v holds K=2v,2v+1;
// lane 16-31 M=lane-16, VGPR v holds K=16+2v,17+2v (matches B packing).
__device__ __forceinline__ void load_afrag(AF& a, const unsigned* astRow,
                                           int m, int g) {
#pragma unroll
  for (int v = 0; v < 8; ++v) {
    int Koff = 16 * g + 2 * v;
    a.u[v] = astRow[m * 16 + (Koff >> 1)];
  }
}

// Write one lane's 16 bf16 A-values (packed in 8 u32) as two b128 LDS stores.
__device__ __forceinline__ void store_alane(unsigned* astRow, int m, int g,
                                            const unsigned au[8]) {
  uint4* ad = (uint4*)&astRow[m * 16 + g * 8];
  ad[0] = make_uint4(au[0], au[1], au[2], au[3]);
  ad[1] = make_uint4(au[4], au[5], au[6], au[7]);
}

// ---------------------------------------------------------------------------
// Kernel 1: offset + modulator conv as im2col GEMM (M=32768, N=32, K=1152).
// WG = 8 waves; wave = 16 positions (one row segment). grid = B * H/2 = 256.
// ---------------------------------------------------------------------------
__global__ __launch_bounds__(256) void offmod_kernel(
    const float* __restrict__ x, const unsigned* __restrict__ woffB,
    const float* __restrict__ b_off, const float* __restrict__ b_mod,
    float* __restrict__ off_out, float* __restrict__ mask_out)
{
  __shared__ __align__(16) unsigned Bst[2 * 32 * 8];    // 2 KB
  __shared__ __align__(16) unsigned Ast[8][256];        // 8 KB (16x32 bf16/wave)

  int tid = threadIdx.x, wv = tid >> 5, lane = tid & 31;
  int b = blockIdx.x >> 5;
  int y = (blockIdx.x & 31) * 2 + (wv >> 2);
  int colbase = (wv & 3) * 16;
  int m = lane & 15, g = lane >> 4;
  int xp = colbase + m;
  const float* xb = x + (size_t)b * Cc * HW;

  v8f zero = {0.f,0.f,0.f,0.f,0.f,0.f,0.f,0.f};
  v8f acc0 = zero, acc1 = zero;

  for (int kc = 0; kc < NKC; ++kc) {
    int t = kc >> 2, cb = (kc & 3) * 32;
    int ty = t / 3 - 1, tx = t % 3 - 1;
    // stage B (512 u32)
    {
      const unsigned* src = woffB + kc * 512;
      Bst[tid]       = src[tid];
      Bst[tid + 256] = src[tid + 256];
    }
    // build A (zero-padded 3x3 window): clamp address, fold validity into a
    // scale factor -> unconditional loads, no exec-mask branches.
    int yy = y + ty, xx = xp + tx;
    bool vrow = (yy >= 0) & (yy < Hh) & (xx >= 0) & (xx < Ww);
    float sc = vrow ? 1.f : 0.f;
    int yyc = min(max(yy, 0), Hh - 1), xxc = min(max(xx, 0), Ww - 1);
    const float* xrow = xb + (size_t)(cb + g * 16) * HW + yyc * Ww + xxc;
    unsigned au[8];
#pragma unroll
    for (int j = 0; j < 8; ++j) {
      float v0 = xrow[(size_t)(2 * j) * HW]     * sc;
      float v1 = xrow[(size_t)(2 * j + 1) * HW] * sc;
      au[j] = pack_bf16(v0, v1);
    }
    store_alane(&Ast[wv][0], m, g, au);
    __syncthreads();
    AF a; load_afrag(a, &Ast[wv][0], m, g);
#pragma unroll
    for (int nt = 0; nt < 2; ++nt) {
      AF bf;
#pragma unroll
      for (int v = 0; v < 8; ++v) bf.u[v] = Bst[(nt * 32 + lane) * 8 + v];
      if (nt == 0)
        acc0 = __builtin_amdgcn_wmma_f32_16x16x32_bf16(false, a.v, false, bf.v,
                                                       (short)0, acc0, false, false);
      else
        acc1 = __builtin_amdgcn_wmma_f32_16x16x32_bf16(false, a.v, false, bf.v,
                                                       (short)0, acc1, false, false);
    }
    __syncthreads();
  }

  // epilogue: +bias, tanh*3 -> offset, sigmoid -> mask (channels 27-31 dropped)
#pragma unroll
  for (int nt = 0; nt < 2; ++nt) {
    int ch = nt * 16 + (lane & 15);
    v8f A = nt ? acc1 : acc0;
#pragma unroll
    for (int r = 0; r < 8; ++r) {
      int M = r + 8 * (lane >> 4);
      int xo = colbase + M;
      float vval = A[r];
      if (ch < 18) {
        float o = tanhf(vval + b_off[ch]) * 3.0f;
        off_out[((size_t)(b * 18 + ch)) * HW + y * Ww + xo] = o;
      } else if (ch < 27) {
        float s = 1.0f / (1.0f + __expf(-(vval + b_mod[ch - 18])));
        mask_out[((size_t)(b * 9 + (ch - 18))) * HW + y * Ww + xo] = s;
      }
    }
  }
}

// ---------------------------------------------------------------------------
// Kernel 2: deformable conv GEMM (M=32768, N=256, K=1152) + BN + exact GELU.
// K-chunk = (tap t, 32-channel block). A built by bilinear gather * mask.
// ---------------------------------------------------------------------------
__global__ __launch_bounds__(256) void deform_kernel(
    const float* __restrict__ x, const unsigned* __restrict__ wregB,
    const float* __restrict__ offs, const float* __restrict__ mask,
    const float* __restrict__ bnp, float* __restrict__ out)
{
  __shared__ __align__(16) unsigned Bst[16 * 32 * 8];   // 16 KB
  __shared__ __align__(16) unsigned Ast[8][256];        // 8 KB
  __shared__ __align__(16) float    Fst[8][256];        // 8 KB (epilogue transpose)

  int tid = threadIdx.x, wv = tid >> 5, lane = tid & 31;
  int b = blockIdx.x >> 5;
  int y = (blockIdx.x & 31) * 2 + (wv >> 2);
  int colbase = (wv & 3) * 16;
  int m = lane & 15, g = lane >> 4;
  int xp = colbase + m;
  const float* xb = x + (size_t)b * Cc * HW;

  v8f zero = {0.f,0.f,0.f,0.f,0.f,0.f,0.f,0.f};
  v8f acc[16];
#pragma unroll
  for (int i = 0; i < 16; ++i) acc[i] = zero;

  for (int kc = 0; kc < NKC; ++kc) {
    int t = kc >> 2, cb = (kc & 3) * 32;
    // prefetch next chunk's B tile while we gather (global_prefetch_b8)
    if (kc + 1 < NKC)
      __builtin_prefetch(wregB + (size_t)(kc + 1) * 4096 + tid * 16, 0, 1);
    // stage B tile (async direct-to-LDS when available)
    stage_b_tile(wregB + (size_t)kc * 4096, Bst, tid);

    // bilinear coords (per position/tap, shared by all channels).
    // Validity is folded into the corner weights; addresses are clamped so
    // all 4*16 channel loads below are unconditional (no exec-mask branches).
    int ky = t / 3 - 1, kx = t % 3 - 1;
    size_t pidx = (size_t)y * Ww + xp;
    float oy = offs[((size_t)(b * 18 + 2 * t)) * HW + pidx];
    float ox = offs[((size_t)(b * 18 + 2 * t + 1)) * HW + pidx];
    float mv = mask[((size_t)(b * 9 + t)) * HW + pidx];
    float py = (float)(y + ky) + oy;
    float px = (float)(xp + kx) + ox;
    float fy = floorf(py), fx = floorf(px);
    float wy = py - fy,    wx = px - fx;
    int y0 = (int)fy, x0 = (int)fx, y1 = y0 + 1, x1 = x0 + 1;
    float vy0 = (y0 >= 0 && y0 < Hh) ? 1.f : 0.f;
    float vy1 = (y1 >= 0 && y1 < Hh) ? 1.f : 0.f;
    float vx0 = (x0 >= 0 && x0 < Ww) ? 1.f : 0.f;
    float vx1 = (x1 >= 0 && x1 < Ww) ? 1.f : 0.f;
    int yc0 = min(max(y0, 0), Hh - 1), yc1 = min(max(y1, 0), Hh - 1);
    int xc0 = min(max(x0, 0), Ww - 1), xc1 = min(max(x1, 0), Ww - 1);
    float w00 = (1.f - wy) * (1.f - wx) * mv * vy0 * vx0;
    float w01 = (1.f - wy) * wx         * mv * vy0 * vx1;
    float w10 = wy * (1.f - wx)         * mv * vy1 * vx0;
    float w11 = wy * wx                 * mv * vy1 * vx1;
    int i00 = yc0 * Ww + xc0, i01 = yc0 * Ww + xc1;
    int i10 = yc1 * Ww + xc0, i11 = yc1 * Ww + xc1;

    const float* xc = xb + (size_t)(cb + g * 16) * HW;
    unsigned au[8];
#pragma unroll
    for (int j = 0; j < 8; ++j) {
      const float* p0 = xc + (size_t)(2 * j) * HW;
      const float* p1 = xc + (size_t)(2 * j + 1) * HW;
      float va = p0[i00] * w00 + p0[i01] * w01 + p0[i10] * w10 + p0[i11] * w11;
      float vb = p1[i00] * w00 + p1[i01] * w01 + p1[i10] * w10 + p1[i11] * w11;
      au[j] = pack_bf16(va, vb);
    }
    store_alane(&Ast[wv][0], m, g, au);
    __syncthreads();

    AF a; load_afrag(a, &Ast[wv][0], m, g);
#pragma unroll
    for (int nt = 0; nt < 16; ++nt) {
      AF bf;
#pragma unroll
      for (int v = 0; v < 8; ++v) bf.u[v] = Bst[(nt * 32 + lane) * 8 + v];
      acc[nt] = __builtin_amdgcn_wmma_f32_16x16x32_bf16(false, a.v, false, bf.v,
                                                        (short)0, acc[nt], false, false);
    }
    __syncthreads();
  }

  // epilogue: BN fold + exact GELU, LDS transpose -> coalesced float4 stores
  const float* s1 = bnp;
  const float* t1 = bnp + 256;
  int chl = lane & 15, half = lane >> 4;
#pragma unroll
  for (int nt = 0; nt < 16; ++nt) {
    int ch = nt * 16 + chl;
    float s = s1[ch], tt = t1[ch];
#pragma unroll
    for (int r = 0; r < 8; ++r) {
      float vv = acc[nt][r] * s + tt;
      vv = 0.5f * vv * (1.0f + erff(vv * 0.70710678118654752f));
      int M = r + 8 * half;
      Fst[wv][M * 16 + chl] = vv;
    }
    __syncthreads();
    float* op = out + ((size_t)(b * Oo + ch)) * HW + (size_t)y * Ww + colbase + half * 8;
    float4 o0, o1;
    o0.x = Fst[wv][(half * 8 + 0) * 16 + chl];
    o0.y = Fst[wv][(half * 8 + 1) * 16 + chl];
    o0.z = Fst[wv][(half * 8 + 2) * 16 + chl];
    o0.w = Fst[wv][(half * 8 + 3) * 16 + chl];
    o1.x = Fst[wv][(half * 8 + 4) * 16 + chl];
    o1.y = Fst[wv][(half * 8 + 5) * 16 + chl];
    o1.z = Fst[wv][(half * 8 + 6) * 16 + chl];
    o1.w = Fst[wv][(half * 8 + 7) * 16 + chl];
    *(float4*)op       = o0;
    *((float4*)op + 1) = o1;
    __syncthreads();
  }
}

// ---------------------------------------------------------------------------
// Kernel 3: 1x1 shortcut GEMM (M=32768, N=256, K=128) + BN, += into d_out.
// ---------------------------------------------------------------------------
__global__ __launch_bounds__(256) void shortcut_kernel(
    const float* __restrict__ x, const unsigned* __restrict__ wscB,
    const float* __restrict__ bnp, float* __restrict__ out)
{
  __shared__ __align__(16) unsigned Bst[16 * 32 * 8];
  __shared__ __align__(16) unsigned Ast[8][256];
  __shared__ __align__(16) float    Fst[8][256];

  int tid = threadIdx.x, wv = tid >> 5, lane = tid & 31;
  int b = blockIdx.x >> 5;
  int y = (blockIdx.x & 31) * 2 + (wv >> 2);
  int colbase = (wv & 3) * 16;
  int m = lane & 15, g = lane >> 4;
  int xp = colbase + m;
  const float* xb = x + (size_t)b * Cc * HW;

  v8f zero = {0.f,0.f,0.f,0.f,0.f,0.f,0.f,0.f};
  v8f acc[16];
#pragma unroll
  for (int i = 0; i < 16; ++i) acc[i] = zero;

  for (int kc = 0; kc < 4; ++kc) {
    if (kc + 1 < 4)
      __builtin_prefetch(wscB + (size_t)(kc + 1) * 4096 + tid * 16, 0, 1);
    stage_b_tile(wscB + (size_t)kc * 4096, Bst, tid);

    const float* xc = xb + (size_t)(kc * 32 + g * 16) * HW + (size_t)y * Ww + xp;
    unsigned au[8];
#pragma unroll
    for (int j = 0; j < 8; ++j) {
      float v0 = xc[(size_t)(2 * j) * HW];
      float v1 = xc[(size_t)(2 * j + 1) * HW];
      au[j] = pack_bf16(v0, v1);
    }
    store_alane(&Ast[wv][0], m, g, au);
    __syncthreads();

    AF a; load_afrag(a, &Ast[wv][0], m, g);
#pragma unroll
    for (int nt = 0; nt < 16; ++nt) {
      AF bf;
#pragma unroll
      for (int v = 0; v < 8; ++v) bf.u[v] = Bst[(nt * 32 + lane) * 8 + v];
      acc[nt] = __builtin_amdgcn_wmma_f32_16x16x32_bf16(false, a.v, false, bf.v,
                                                        (short)0, acc[nt], false, false);
    }
    __syncthreads();
  }

  const float* s2 = bnp + 512;
  const float* t2 = bnp + 768;
  int chl = lane & 15, half = lane >> 4;
#pragma unroll
  for (int nt = 0; nt < 16; ++nt) {
    int ch = nt * 16 + chl;
    float s = s2[ch], tt = t2[ch];
#pragma unroll
    for (int r = 0; r < 8; ++r) {
      float vv = acc[nt][r] * s + tt;
      int M = r + 8 * half;
      Fst[wv][M * 16 + chl] = vv;
    }
    __syncthreads();
    float* op = out + ((size_t)(b * Oo + ch)) * HW + (size_t)y * Ww + colbase + half * 8;
    float4 c0 = *(float4*)op;
    float4 c1 = *((float4*)op + 1);
    c0.x += Fst[wv][(half * 8 + 0) * 16 + chl];
    c0.y += Fst[wv][(half * 8 + 1) * 16 + chl];
    c0.z += Fst[wv][(half * 8 + 2) * 16 + chl];
    c0.w += Fst[wv][(half * 8 + 3) * 16 + chl];
    c1.x += Fst[wv][(half * 8 + 4) * 16 + chl];
    c1.y += Fst[wv][(half * 8 + 5) * 16 + chl];
    c1.z += Fst[wv][(half * 8 + 6) * 16 + chl];
    c1.w += Fst[wv][(half * 8 + 7) * 16 + chl];
    *(float4*)op       = c0;
    *((float4*)op + 1) = c1;
    __syncthreads();
  }
}

// ---------------------------------------------------------------------------
extern "C" void kernel_launch(void* const* d_in, const int* in_sizes, int n_in,
                              void* d_out, int out_size, void* d_ws, size_t ws_size,
                              hipStream_t stream)
{
  const float* x     = (const float*)d_in[0];
  const float* w_off = (const float*)d_in[1];
  const float* b_off = (const float*)d_in[2];
  const float* w_mod = (const float*)d_in[3];
  const float* b_mod = (const float*)d_in[4];
  const float* w_reg = (const float*)d_in[5];
  const float* b_reg = (const float*)d_in[6];
  const float* g1  = (const float*)d_in[7];
  const float* be1 = (const float*)d_in[8];
  const float* mu1 = (const float*)d_in[9];
  const float* va1 = (const float*)d_in[10];
  const float* sc_w = (const float*)d_in[11];
  const float* g2  = (const float*)d_in[12];
  const float* be2 = (const float*)d_in[13];
  const float* mu2 = (const float*)d_in[14];
  const float* va2 = (const float*)d_in[15];
  float* out = (float*)d_out;

  // workspace partition (≈ 4.3 MB)
  float*    off   = (float*)d_ws;                    // 589824 f32
  float*    maskp = off + 589824;                    // 294912 f32
  unsigned* wregB = (unsigned*)(maskp + 294912);     // 147456 u32
  unsigned* woffB = wregB + 147456;                  // 18432 u32
  unsigned* wscB  = woffB + 18432;                   // 16384 u32
  float*    bnp   = (float*)(wscB + 16384);          // 1024 f32

  const int total = 147456 + 18432 + 16384;
  prep_kernel<<<(total + 255) / 256, 256, 0, stream>>>(
      w_off, w_mod, w_reg, sc_w, b_reg,
      g1, be1, mu1, va1, g2, be2, mu2, va2,
      wregB, woffB, wscB, bnp);

  offmod_kernel<<<256, 256, 0, stream>>>(x, woffB, b_off, b_mod, off, maskp);
  deform_kernel<<<256, 256, 0, stream>>>(x, wregB, off, maskp, bnp, out);
  shortcut_kernel<<<256, 256, 0, stream>>>(x, wscB, bnp, out);
}
